// Attention_dual_1606317768801
// MI455X (gfx1250) — matrix operand order repaired and checked by reading, not compiled, not run on hardware
//
#include <hip/hip_runtime.h>
#include <hip/hip_bf16.h>
#include <math.h>

typedef __attribute__((ext_vector_type(16))) _Float16 v16h;
typedef __attribute__((ext_vector_type(4)))  _Float16 v4h;
typedef __attribute__((ext_vector_type(8)))  float    v8f;
typedef __attribute__((ext_vector_type(4)))  float    v4f;

#define NINST 64          // B(8) * NUM_HEADS(8)
#define SPAT  16384       // H*W
#define CH    64          // IN_C == C_M == C_N
#define LDH   72          // LDS f16 row stride (halves); 144B rows, 8B aligned cols
#define LDZ   68          // LDS f32 row stride for Z staging; 272B rows (16B aligned)

#define NS1 64            // kernel1 spatial splits (span 256)
#define NS2 32            // kernel2 spatial splits (span 512)
#define NS3 64            // kernel3 spatial splits (span 256)

// scratch layout (float units)
#define RM_OFF  0
#define DEN_OFF (NINST*CH)              // 4096
#define Y_OFF   (2*NINST*CH)            // 8192
#define G_OFF   (Y_OFF + NINST*CH*CH)   // 8192 + 262144
#define ZERO_N  G_OFF                   // zero rowmax/den/Y regions

static __device__ inline v8f wmma_f16(v16h a, v16h b, v8f c) {
  // D(16x16,f32) = A(16x32,f16) x B(32x16,f16) + C
  return __builtin_amdgcn_wmma_f32_16x16x32_f16(false, a, false, b, (short)0, c,
                                                false, false);
}

// A operand (16M x 32K, f16) from LDS stored row-major [m][k], stride LDH.
// ISA 7.12.2: lane<16 -> M=m0+lane, K in {k0..k0+7, k0+16..k0+23};
//             lane>=16 -> same M set, K shifted by +8.
static __device__ inline v16h frag_a(const _Float16* lds, int m0, int k0) {
  const int lane = threadIdx.x & 31;
  const int m  = m0 + (lane & 15);
  const int kb = k0 + ((lane >> 4) << 3);
  const _Float16* p = lds + m * LDH + kb;
  v16h r;
#pragma unroll
  for (int i = 0; i < 8; ++i) r[i] = p[i];
#pragma unroll
  for (int i = 0; i < 8; ++i) r[8 + i] = p[16 + i];
  return r;
}

// B operand (32K x 16N, f16) from LDS stored [n][k], stride LDH.
// lane<16 -> N=n0+lane, K in {k0..k0+7, k0+16..k0+23}; lane>=16 -> K shifted by +8.
static __device__ inline v16h frag_b(const _Float16* lds, int k0, int n0) {
  const int lane = threadIdx.x & 31;
  const int n  = n0 + (lane & 15);
  const int kb = k0 + ((lane >> 4) << 3);
  const _Float16* p = lds + n * LDH + kb;
  v16h r;
#pragma unroll
  for (int i = 0; i < 8; ++i) r[i] = p[i];
#pragma unroll
  for (int i = 0; i < 8; ++i) r[8 + i] = p[16 + i];
  return r;
}

// 64x64 f32 matrix (row-major, contiguous) -> f16 LDS [m][k] stride LDH.
// Phase 1: 8x global_load_b128 batched; Phase 2: cvt + ds_store_b64.
static __device__ inline void ld_mat64(_Float16* dst, const float* __restrict__ src) {
  const int tid = threadIdx.x;
  v4f t[8];
#pragma unroll
  for (int g = 0; g < 8; ++g) t[g] = *(const v4f*)(src + (g * 128 + tid) * 4);
#pragma unroll
  for (int g = 0; g < 8; ++g) {
    const int idx4 = g * 128 + tid;
    const int m = idx4 >> 4, k4 = idx4 & 15;
    v4h h;
#pragma unroll
    for (int r = 0; r < 4; ++r) h[r] = (_Float16)t[g][r];
    *(v4h*)(dst + m * LDH + k4 * 4) = h;
  }
}

// 64(ch) x 64(s) f32 tile of x -> LDS, transposed sXT[s][ch] (+ optional sXK[ch][s]).
// Each thread owns column j and 4-row strips, so the transposed store is a
// contiguous ds_store_b64. Loads are issued in a batch (32 in flight), then
// converted — avoids per-element s_wait_loadcnt.
template <bool DUAL>
static __device__ inline void ld_xtile(const float* __restrict__ xg, int c0,
                                       _Float16* sXT, _Float16* sXK) {
  const int tid = threadIdx.x;
  const int j  = tid & 63;
  const int qb = (tid >> 6) * 4;   // 0 or 4
  float t[8][4];
#pragma unroll
  for (int g = 0; g < 8; ++g) {
    const int k0 = g * 8 + qb;
#pragma unroll
    for (int r = 0; r < 4; ++r)
      t[g][r] = xg[(size_t)(k0 + r) * SPAT + c0 + j];
  }
#pragma unroll
  for (int g = 0; g < 8; ++g) {
    const int k0 = g * 8 + qb;
    v4h h;
#pragma unroll
    for (int r = 0; r < 4; ++r) h[r] = (_Float16)t[g][r];
    *(v4h*)(sXT + j * LDH + k0) = h;   // contiguous in k along the strip
    if (DUAL) {
#pragma unroll
      for (int r = 0; r < 4; ++r) sXK[(k0 + r) * LDH + j] = h[r];
    }
  }
}

// monotone float<->uint map for atomicMax on floats
static __device__ inline unsigned fenc(float f) {
  unsigned u = __float_as_uint(f);
  return (u & 0x80000000u) ? ~u : (u | 0x80000000u);
}
static __device__ inline float fdec(unsigned e) {
  unsigned u = (e & 0x80000000u) ? (e ^ 0x80000000u) : ~e;
  return __uint_as_float(u);
}

__global__ void k0_init(float* __restrict__ ws, int n) {
  int i = blockIdx.x * blockDim.x + threadIdx.x;
  if (i < n) ws[i] = 0.0f;
}

// ---------------- kernel 1: rowmax_n over S of Bm = Wb @ x ----------------
__global__ void __launch_bounds__(128)
k1_rowmax(const float* __restrict__ x, const float* __restrict__ Wb,
          unsigned* __restrict__ rm) {
  __shared__ _Float16 sW[CH * LDH];
  __shared__ _Float16 sXT[64 * LDH];   // [s][channel]
  const int inst = blockIdx.y;
  const int tid = threadIdx.x, wave = tid >> 5, lane = tid & 31;
  const int hi8 = (lane >> 4) << 3;
  const float* xg = x + (size_t)inst * CH * SPAT;

  ld_mat64(sW, Wb);
  __syncthreads();

  v16h wf[4][2];
#pragma unroll
  for (int nb = 0; nb < 4; ++nb)
#pragma unroll
    for (int kb = 0; kb < 2; ++kb) wf[nb][kb] = frag_a(sW, nb * 16, kb * 32);

  float rmax[4][8];
#pragma unroll
  for (int nb = 0; nb < 4; ++nb)
#pragma unroll
    for (int v = 0; v < 8; ++v) rmax[nb][v] = -3.0e38f;

  const int span = SPAT / NS1;
  const int base = blockIdx.x * span;
  const int iters = span / 64;
  for (int it = 0; it < iters; ++it) {
    const int c0 = base + it * 64;
    __syncthreads();
    ld_xtile<false>(xg, c0, sXT, sXT);
    __syncthreads();
    v16h xb0 = frag_b(sXT, 0, wave * 16);
    v16h xb1 = frag_b(sXT, 32, wave * 16);
#pragma unroll
    for (int nb = 0; nb < 4; ++nb) {
      v8f acc = {};
      acc = wmma_f16(wf[nb][0], xb0, acc);
      acc = wmma_f16(wf[nb][1], xb1, acc);
#pragma unroll
      for (int v = 0; v < 8; ++v) rmax[nb][v] = fmaxf(rmax[nb][v], acc[v]);
    }
  }
#pragma unroll
  for (int m = 1; m <= 8; m <<= 1)
#pragma unroll
    for (int nb = 0; nb < 4; ++nb)
#pragma unroll
      for (int v = 0; v < 8; ++v)
        rmax[nb][v] = fmaxf(rmax[nb][v], __shfl_xor(rmax[nb][v], m, 32));
  if ((lane & 15) == 0) {
#pragma unroll
    for (int nb = 0; nb < 4; ++nb)
#pragma unroll
      for (int v = 0; v < 8; ++v)
        atomicMax(&rm[inst * CH + nb * 16 + v + hi8], fenc(rmax[nb][v]));
  }
}

// -------- kernel 2: den_n = sum_s E[n,s], Y = x @ E^T  (E = exp(Bm-max)) --------
__global__ void __launch_bounds__(128)
k2_yden(const float* __restrict__ x, const float* __restrict__ Wb,
        const unsigned* __restrict__ rm, float* __restrict__ den,
        float* __restrict__ Yb) {
  __shared__ _Float16 sW[CH * LDH];
  __shared__ _Float16 sXT[64 * LDH];   // [s][channel]  (B operand for Bm)
  __shared__ _Float16 sXK[CH * LDH];   // [channel][s]  (A operand for Y)
  __shared__ _Float16 sE[CH * LDH];    // [n][s]        (B operand for Y)
  __shared__ float sMax[CH];
  const int inst = blockIdx.y;
  const int tid = threadIdx.x, wave = tid >> 5, lane = tid & 31;
  const int hi8 = (lane >> 4) << 3;
  const float* xg = x + (size_t)inst * CH * SPAT;

  ld_mat64(sW, Wb);
  if (tid < CH) sMax[tid] = fdec(rm[inst * CH + tid]);
  __syncthreads();

  v16h wf[4][2];
#pragma unroll
  for (int nb = 0; nb < 4; ++nb)
#pragma unroll
    for (int kb = 0; kb < 2; ++kb) wf[nb][kb] = frag_a(sW, nb * 16, kb * 32);

  float mx[4][8];
#pragma unroll
  for (int nb = 0; nb < 4; ++nb)
#pragma unroll
    for (int v = 0; v < 8; ++v) mx[nb][v] = sMax[nb * 16 + v + hi8];

  float dsum[4][8] = {};
  v8f zero = {};
  v8f yacc[4];
#pragma unroll
  for (int nb = 0; nb < 4; ++nb) yacc[nb] = zero;

  const int span = SPAT / NS2;
  const int base = blockIdx.x * span;
  const int iters = span / 64;
  const int scol = wave * 16 + (lane & 15);
  for (int it = 0; it < iters; ++it) {
    const int c0 = base + it * 64;
    __syncthreads();
    ld_xtile<true>(xg, c0, sXT, sXK);
    __syncthreads();
    // stage 1: Bm tiles for this wave's 16 columns; E -> LDS (transposed store)
    v16h xb0 = frag_b(sXT, 0, wave * 16);
    v16h xb1 = frag_b(sXT, 32, wave * 16);
#pragma unroll
    for (int nb = 0; nb < 4; ++nb) {
      v8f acc = {};
      acc = wmma_f16(wf[nb][0], xb0, acc);
      acc = wmma_f16(wf[nb][1], xb1, acc);
#pragma unroll
      for (int v = 0; v < 8; ++v) {
        float e = __expf(acc[v] - mx[nb][v]);
        dsum[nb][v] += e;
        sE[(nb * 16 + v + hi8) * LDH + scol] = (_Float16)e;
      }
    }
    __syncthreads();
    // stage 2: Y[i,n] += x(16i x 64s) @ E^T(64s x 16n); wave owns i-block = wave
    v16h xa0 = frag_a(sXK, wave * 16, 0);
    v16h xa1 = frag_a(sXK, wave * 16, 32);
#pragma unroll
    for (int nb = 0; nb < 4; ++nb) {
      yacc[nb] = wmma_f16(xa0, frag_b(sE, 0, nb * 16), yacc[nb]);
      yacc[nb] = wmma_f16(xa1, frag_b(sE, 32, nb * 16), yacc[nb]);
    }
  }
#pragma unroll
  for (int m = 1; m <= 8; m <<= 1)
#pragma unroll
    for (int nb = 0; nb < 4; ++nb)
#pragma unroll
      for (int v = 0; v < 8; ++v)
        dsum[nb][v] += __shfl_xor(dsum[nb][v], m, 32);
  if ((lane & 15) == 0) {
#pragma unroll
    for (int nb = 0; nb < 4; ++nb)
#pragma unroll
      for (int v = 0; v < 8; ++v)
        atomicAdd(&den[inst * CH + nb * 16 + v + hi8], dsum[nb][v]);
  }
#pragma unroll
  for (int nb = 0; nb < 4; ++nb)
#pragma unroll
    for (int v = 0; v < 8; ++v) {
      int i = wave * 16 + v + hi8;          // D layout: VGPR -> M(+8 for hi lanes)
      int n = nb * 16 + (lane & 15);        // D layout: lane  -> N
      atomicAdd(&Yb[(size_t)inst * CH * CH + i * CH + n], yacc[nb][v]);
    }
}

// ---------------- kernel 3a: G = Wa @ Y / den + ba  (tiny 64^3) ----------------
__global__ void __launch_bounds__(64)
k3a_G(const float* __restrict__ Wa, const float* __restrict__ ba,
      const float* __restrict__ Yb, const float* __restrict__ den,
      float* __restrict__ G) {
  __shared__ float sWa[CH * CH];
  __shared__ float sY[CH * CH];
  __shared__ float sDen[CH];
  const int inst = blockIdx.x, t = threadIdx.x;
  for (int r = 0; r < CH; ++r) {
    sWa[r * CH + t] = Wa[r * CH + t];
    sY[r * CH + t]  = Yb[(size_t)inst * CH * CH + r * CH + t];
  }
  sDen[t] = den[inst * CH + t];
  __syncthreads();
  const int n = t;
  const float invd = 1.0f / sDen[n];
  for (int m = 0; m < CH; ++m) {
    float s = 0.0f;
#pragma unroll 8
    for (int i = 0; i < CH; ++i) s += sWa[m * CH + i] * sY[i * CH + n];
    G[(size_t)inst * CH * CH + m * CH + n] = ba[m] + s * invd;
  }
}

// -------- kernel 3b: V = Wv@x + bv ; channel-softmax ; Z = G @ attn ; write --------
__global__ void __launch_bounds__(128)
k3b_out(const float* __restrict__ x, const float* __restrict__ Wv,
        const float* __restrict__ bv, const float* __restrict__ G,
        float* __restrict__ out) {
  __shared__ _Float16 sW[CH * LDH];
  __shared__ _Float16 sG[CH * LDH];    // [m][n] for A operand of Z GEMM
  __shared__ _Float16 sXT[64 * LDH];   // [s][channel]
  __shared__ _Float16 sAT[64 * LDH];   // [s][n] for B operand of Z GEMM
  __shared__ float sZ[CH * LDZ];
  __shared__ float sBv[CH];
  const int inst = blockIdx.y;
  const int tid = threadIdx.x, wave = tid >> 5, lane = tid & 31;
  const int hi8 = (lane >> 4) << 3;
  const float* xg = x + (size_t)inst * CH * SPAT;
  float* og = out + (size_t)inst * CH * SPAT;

  ld_mat64(sW, Wv);
  ld_mat64(sG, G + (size_t)inst * CH * CH);
  if (tid < CH) sBv[tid] = bv[tid];
  __syncthreads();

  v16h wf[4][2], gf[4][2];
#pragma unroll
  for (int b = 0; b < 4; ++b)
#pragma unroll
    for (int kb = 0; kb < 2; ++kb) {
      wf[b][kb] = frag_a(sW, b * 16, kb * 32);
      gf[b][kb] = frag_a(sG, b * 16, kb * 32);
    }
  float bvr[4][8];
#pragma unroll
  for (int nb = 0; nb < 4; ++nb)
#pragma unroll
    for (int v = 0; v < 8; ++v) bvr[nb][v] = sBv[nb * 16 + v + hi8];

  const int span = SPAT / NS3;
  const int base = blockIdx.x * span;
  const int iters = span / 64;
  const int scol = wave * 16 + (lane & 15);
  for (int it = 0; it < iters; ++it) {
    const int c0 = base + it * 64;
    __syncthreads();
    ld_xtile<false>(xg, c0, sXT, sXT);
    __syncthreads();
    // V tiles for this wave's 16 columns
    v16h xb0 = frag_b(sXT, 0, wave * 16);
    v16h xb1 = frag_b(sXT, 32, wave * 16);
    float vals[4][8];
#pragma unroll
    for (int nb = 0; nb < 4; ++nb) {
      v8f acc = {};
      acc = wmma_f16(wf[nb][0], xb0, acc);
      acc = wmma_f16(wf[nb][1], xb1, acc);
#pragma unroll
      for (int v = 0; v < 8; ++v) vals[nb][v] = acc[v] + bvr[nb][v];
    }
    // softmax over the 64 channels of this column (lane holds 32, partner lane+16 rest)
    float m1 = -3.0e38f;
#pragma unroll
    for (int nb = 0; nb < 4; ++nb)
#pragma unroll
      for (int v = 0; v < 8; ++v) m1 = fmaxf(m1, vals[nb][v]);
    m1 = fmaxf(m1, __shfl_xor(m1, 16, 32));
    float ssum = 0.0f, ex[4][8];
#pragma unroll
    for (int nb = 0; nb < 4; ++nb)
#pragma unroll
      for (int v = 0; v < 8; ++v) {
        ex[nb][v] = __expf(vals[nb][v] - m1);
        ssum += ex[nb][v];
      }
    ssum += __shfl_xor(ssum, 16, 32);
    const float inv = 1.0f / ssum;
#pragma unroll
    for (int nb = 0; nb < 4; ++nb)
#pragma unroll
      for (int v = 0; v < 8; ++v)
        sAT[scol * LDH + nb * 16 + hi8 + v] = (_Float16)(ex[nb][v] * inv);
    __syncthreads();
    // Z = G(64x64) @ attn(64 x 16s)
    v16h ab0 = frag_b(sAT, 0, wave * 16);
    v16h ab1 = frag_b(sAT, 32, wave * 16);
#pragma unroll
    for (int mb = 0; mb < 4; ++mb) {
      v8f z = {};
      z = wmma_f16(gf[mb][0], ab0, z);
      z = wmma_f16(gf[mb][1], ab1, z);
#pragma unroll
      for (int v = 0; v < 8; ++v)
        sZ[(mb * 16 + v + hi8) * LDZ + scol] = z[v];
    }
    __syncthreads();
    // coalesced b128 store to HBM
#pragma unroll
    for (int g = 0; g < 8; ++g) {
      const int idx4 = g * 128 + tid;
      const int m = idx4 >> 4, j4 = idx4 & 15;
      *(v4f*)(og + (size_t)m * SPAT + c0 + j4 * 4) =
          *(const v4f*)(sZ + m * LDZ + j4 * 4);
    }
  }
}

extern "C" void kernel_launch(void* const* d_in, const int* in_sizes, int n_in,
                              void* d_out, int out_size, void* d_ws, size_t ws_size,
                              hipStream_t stream) {
  (void)in_sizes; (void)n_in; (void)out_size; (void)ws_size;
  const float* x  = (const float*)d_in[0];
  const float* Wa = (const float*)d_in[1];
  const float* ba = (const float*)d_in[2];
  const float* Wb = (const float*)d_in[3];
  // d_in[4] = bb: constant along the softmax(S) axis -> cancels; unused.
  const float* Wv = (const float*)d_in[5];
  const float* bv = (const float*)d_in[6];
  float* out = (float*)d_out;
  float* ws  = (float*)d_ws;

  unsigned* rm = (unsigned*)(ws + RM_OFF);
  float* den = ws + DEN_OFF;
  float* Yb  = ws + Y_OFF;
  float* G   = ws + G_OFF;

  k0_init<<<(ZERO_N + 255) / 256, 256, 0, stream>>>(ws, ZERO_N);
  k1_rowmax<<<dim3(NS1, NINST), 128, 0, stream>>>(x, Wb, rm);
  k2_yden<<<dim3(NS2, NINST), 128, 0, stream>>>(x, Wb, rm, den, Yb);
  k3a_G<<<NINST, 64, 0, stream>>>(Wa, ba, Yb, den, G);
  k3b_out<<<dim3(NS3, NINST), 128, 0, stream>>>(x, Wv, bv, G, out);
}
